// CustomLSTM_13099650253140
// MI455X (gfx1250) — matrix operand order repaired and checked by reading, not compiled, not run on hardware
//
#include <hip/hip_runtime.h>
#include <stdint.h>

typedef __attribute__((ext_vector_type(16))) __bf16 v16bf;
typedef __attribute__((ext_vector_type(8)))  __bf16 v8bf;
typedef __attribute__((ext_vector_type(8)))  float  v8f;

// ---------- helpers ----------
__device__ __forceinline__ uint16_t f2bf(float x) {
    uint32_t u = __float_as_uint(x);
    uint32_t r = u + 0x7FFFu + ((u >> 16) & 1u);   // round-to-nearest-even
    return (uint16_t)(r >> 16);
}
__device__ __forceinline__ float sigmoidf_(float x) {
    return 1.0f / (1.0f + __expf(-x));
}

// ---------- prep kernels ----------
__global__ void k_convert_x(const float* __restrict__ x, uint16_t* __restrict__ xb, int n) {
    int i = blockIdx.x * blockDim.x + threadIdx.x;
    if (i < n) xb[i] = f2bf(x[i]);
}

// Wt[n][k] = bf16(W[k][n]);  W is KxN row-major. LDS-tiled for coalescing.
__global__ __launch_bounds__(256) void k_transpose_bf16(const float* __restrict__ W,
                                                        uint16_t* __restrict__ Wt,
                                                        int K, int N) {
    __shared__ float tile[32][33];
    int nb = blockIdx.x * 32;
    int kb = blockIdx.y * 32;
    for (int dy = threadIdx.y; dy < 32; dy += 8)
        tile[dy][threadIdx.x] = W[(size_t)(kb + dy) * N + (nb + threadIdx.x)];
    __syncthreads();
    for (int dy = threadIdx.y; dy < 32; dy += 8)
        Wt[(size_t)(nb + dy) * K + (kb + threadIdx.x)] = f2bf(tile[threadIdx.x][dy]);
}

__global__ void k_gather_bias(const float* __restrict__ b0, const float* __restrict__ b1,
                              const float* __restrict__ b2, const float* __restrict__ b3,
                              float* __restrict__ out) {
    int i = blockIdx.x * blockDim.x + threadIdx.x;
    if (i < 1024) {
        out[i] = b0[i]; out[1024 + i] = b1[i]; out[2048 + i] = b2[i]; out[3072 + i] = b3[i];
    }
}

__global__ void k_zero_state(float* __restrict__ c, uint16_t* __restrict__ h0,
                             uint16_t* __restrict__ h1, int n) {
    int i = blockIdx.x * blockDim.x + threadIdx.x;
    if (i < n) { c[i] = 0.0f; h0[i] = 0; h1[i] = 0; }
}

// K-chunk WMMA helper: one 16x16x32 bf16 MAC from row pointers.
__device__ __forceinline__ v8f wmma_chunk(const uint16_t* __restrict__ arow,
                                          const uint16_t* __restrict__ brow,
                                          int kk, int kA, int kB, v8f acc) {
    v8bf a0 = *(const v8bf*)(arow + kk + kA);
    v8bf a1 = *(const v8bf*)(arow + kk + kA + 8);
    v8bf b0 = *(const v8bf*)(brow + kk + kB);
    v8bf b1 = *(const v8bf*)(brow + kk + kB + 8);
    v16bf A, B;
#pragma unroll
    for (int j = 0; j < 8; ++j) { A[j] = a0[j]; A[j + 8] = a1[j]; }
#pragma unroll
    for (int j = 0; j < 8; ++j) { B[j] = b0[j]; B[j + 8] = b1[j]; }
    return __builtin_amdgcn_wmma_f32_16x16x32_bf16(false, A, false, B,
                                                   (short)0, acc, false, false);
}

// ---------- fused LSTM step ----------
// Grid: 256 blocks = 4 m-tiles (batch) x 64 n-tiles (hidden). Block: 128 thr = 4 waves.
// Wave g computes gate g's 16x16 pre-activation tile with TWO independent
// accumulator chains (even/odd K chunks) for matrix-pipe ILP:
//   acc = bias_g[n] + x[t] @ W2_g  (K=512)  + h @ W1_g  (K=1024)
// then gates are exchanged via LDS and the elementwise update runs fused.
__global__ __launch_bounds__(128) void k_lstm_step(
    const uint16_t* __restrict__ xbf,   // [64][512][512] bf16
    const uint16_t* __restrict__ wt1,   // [4][1024 n][1024 k] bf16 (recurrent, transposed)
    const uint16_t* __restrict__ wt2,   // [4][1024 n][512 k]  bf16 (input, transposed)
    const float*    __restrict__ bias,  // [4][1024]
    float*          __restrict__ c,     // [64][1024] fp32, in-place
    const uint16_t* __restrict__ hin,   // [64][1024] bf16
    uint16_t*       __restrict__ hout,  // [64][1024] bf16 (double buffer)
    float*          __restrict__ hf32,  // [64][1024] fp32 (d_out)
    int t)
{
    __shared__ float lds[4][16][16];

    const int gate = threadIdx.x >> 5;
    const int lane = threadIdx.x & 31;
    const int mt   = blockIdx.x & 3;     // batch tile (64/16)
    const int nt   = blockIdx.x >> 2;    // hidden tile (1024/16)
    const int mrow = lane & 15;          // A-matrix row within tile
    const int half = lane >> 4;          // lane group 0/1
    const int gn   = nt * 16 + (lane & 15);  // B-matrix column (hidden unit)
    const int kA   = half * 16;          // A K-offset, contiguous k = 16*half + s
    const int kB   = half * 16;          // B K-offset, contiguous k = 16*half + s

    // two independent accumulator chains; bias goes in chain 0 only
    const float bval = bias[gate * 1024 + gn];
    v8f acc0, acc1;
#pragma unroll
    for (int r = 0; r < 8; ++r) { acc0[r] = bval; acc1[r] = 0.0f; }

    // ---- x-projection part: A = x[:, t, :], K = 512 ----
    {
        const uint16_t* arow = xbf + ((size_t)(mt * 16 + mrow) * 512 + (size_t)t) * 512;
        const uint16_t* brow = wt2 + ((size_t)gate * 1024 + gn) * 512;
#pragma unroll 2
        for (int kk = 0; kk < 512; kk += 64) {
            acc0 = wmma_chunk(arow, brow, kk,      kA, kB, acc0);
            acc1 = wmma_chunk(arow, brow, kk + 32, kA, kB, acc1);
        }
    }

    // ---- recurrent part: A = h_{t-1}, K = 1024 ----
    {
        const uint16_t* arow = hin + (size_t)(mt * 16 + mrow) * 1024;
        const uint16_t* brow = wt1 + ((size_t)gate * 1024 + gn) * 1024;
#pragma unroll 2
        for (int kk = 0; kk < 1024; kk += 64) {
            acc0 = wmma_chunk(arow, brow, kk,      kA, kB, acc0);
            acc1 = wmma_chunk(arow, brow, kk + 32, kA, kB, acc1);
        }
    }

    // ---- merge chains, exchange gate tiles via LDS ----
    // (C/D layout: reg r, lane L -> m=(L>>4)*8+r, n=L&15)
#pragma unroll
    for (int r = 0; r < 8; ++r) lds[gate][half * 8 + r][lane & 15] = acc0[r] + acc1[r];
    __syncthreads();

    // ---- fused elementwise LSTM cell update (256 elems, 128 threads -> 2 each) ----
#pragma unroll
    for (int e = threadIdx.x; e < 256; e += 128) {
        const int m = e >> 4, n = e & 15;
        const float f = sigmoidf_(lds[0][m][n]);
        const float i = sigmoidf_(lds[1][m][n]);
        const float g = tanhf(lds[2][m][n]);
        const float o = sigmoidf_(lds[3][m][n]);
        const size_t idx = (size_t)(mt * 16 + m) * 1024 + (nt * 16 + n);
        const float cn = c[idx] * f + g * i;
        const float hn = tanhf(cn) * o;
        c[idx]    = cn;
        hout[idx] = f2bf(hn);
        hf32[idx] = hn;
    }
}

// ---------- host launcher ----------
extern "C" void kernel_launch(void* const* d_in, const int* in_sizes, int n_in,
                              void* d_out, int out_size, void* d_ws, size_t ws_size,
                              hipStream_t stream) {
    (void)in_sizes; (void)n_in; (void)out_size; (void)ws_size;

    const float* x    = (const float*)d_in[0];
    const float* wlr1 = (const float*)d_in[1];
    const float* wlr2 = (const float*)d_in[2];
    const float* blr1 = (const float*)d_in[3];
    const float* wpr1 = (const float*)d_in[4];
    const float* wpr2 = (const float*)d_in[5];
    const float* bpr1 = (const float*)d_in[6];
    const float* wp1  = (const float*)d_in[7];
    const float* wp2  = (const float*)d_in[8];
    const float* bp1  = (const float*)d_in[9];
    const float* wo1  = (const float*)d_in[10];
    const float* wo2  = (const float*)d_in[11];
    const float* bo1  = (const float*)d_in[12];

    const int B = 64, T = 512, I = 512, H = 1024;

    // workspace carve-up (all offsets 256B-aligned); total ~44.6 MB
    char* ws = (char*)d_ws;
    uint16_t* xbf  = (uint16_t*)(ws);                       // 64*512*512*2  = 32 MB
    uint16_t* wt1  = (uint16_t*)(ws + 33554432);            // 4*1024*1024*2 = 8 MB
    uint16_t* wt2  = (uint16_t*)(ws + 41943040);            // 4*1024*512*2  = 4 MB
    float*    bias = (float*)   (ws + 46137344);            // 4*1024*4
    float*    cst  = (float*)   (ws + 46153728);            // 64*1024*4
    uint16_t* hbf0 = (uint16_t*)(ws + 46415872);            // 64*1024*2
    uint16_t* hbf1 = (uint16_t*)(ws + 46546944);            // 64*1024*2

    // 1) convert x to bf16
    {
        int n = B * T * I;
        k_convert_x<<<(n + 255) / 256, 256, 0, stream>>>(x, xbf, n);
    }
    // 2) transpose + convert weights (gate order: f, i, g, o)
    {
        dim3 blk(32, 8);
        dim3 grR(H / 32, H / 32);   // recurrent K=H, N=H
        dim3 grI(H / 32, I / 32);   // input     K=I, N=H
        const float* wr[4] = { wlr1, wpr1, wp1, wo1 };
        const float* wi[4] = { wlr2, wpr2, wp2, wo2 };
        for (int g = 0; g < 4; ++g) {
            k_transpose_bf16<<<grR, blk, 0, stream>>>(wr[g], wt1 + (size_t)g * H * H, H, H);
            k_transpose_bf16<<<grI, blk, 0, stream>>>(wi[g], wt2 + (size_t)g * H * I, I, H);
        }
    }
    // 3) biases + zero state
    k_gather_bias<<<4, 256, 0, stream>>>(blr1, bpr1, bp1, bo1, bias);
    {
        int n = B * H;
        k_zero_state<<<(n + 255) / 256, 256, 0, stream>>>(cst, hbf0, hbf1, n);
    }
    // 4) sequential scan: 512 fused steps (h double-buffered; c in place)
    float* hf32 = (float*)d_out;   // overwritten every step; last step leaves the answer
    for (int t = 0; t < T; ++t) {
        const uint16_t* hin  = (t & 1) ? hbf1 : hbf0;
        uint16_t*       hout = (t & 1) ? hbf0 : hbf1;
        k_lstm_step<<<256, 128, 0, stream>>>(xbf, wt1, wt2, bias, cst, hin, hout, hf32, t);
    }
}
